// DirectionAttractorNet_14250701488451
// MI455X (gfx1250) — compile-verified, matrix operand order checked by reading, not run on hardware
//
#include <hip/hip_runtime.h>
#include <hip/hip_bf16.h>
#include <math.h>

// ---------------------------------------------------------------- constants
#define PI_F 3.14159265358979f
#define BATCH 4
#define LS    128000
#define NFFT  512
#define HOP   128
#define NH    257          // freq bins
#define NT_   1001         // time frames
#define MBT   (BATCH*NT_)  // 4004 rows
#define MP    4032         // padded rows (mult of 32)
#define KP288 288          // 257 padded to k-mult-32
#define KP512 512
#define KP544 544          // 514 padded
#define NP_DFT  576        // 514 padded to n-mult-64
#define NP_MASK 320        // 257 padded to n-mult-64
#define NP_TIME 512
#define TOTLEN  ((NT_-1)*HOP + NFFT)   // 128512
#define NGATE 771
// GRU partitioning: 16 WGs per chain, each owns 17 h-indices (last owns 2)
#define GRU_NWG  16
#define GRU_JSL  17
#define GRU_ROWS 64        // 3*17=51 gate rows padded to 4 n-tiles
#define GRU_KT   9         // 288/32

typedef __attribute__((ext_vector_type(16))) _Float16 v16h;
typedef __attribute__((ext_vector_type(8)))  _Float16 v8h;
typedef __attribute__((ext_vector_type(8)))  float    v8f;

// ---------------------------------------------------------------- ws layout
constexpr size_t AL(size_t x) { return (x + 255) & ~(size_t)255; }
constexpr size_t OFF_WIN   = 0;
constexpr size_t OFF_DFTT  = AL(OFF_WIN   + 512*4);
constexpr size_t OFF_IDFTT = AL(OFF_DFTT  + (size_t)NP_DFT*512*2);
constexpr size_t OFF_WMS   = AL(OFF_IDFTT + (size_t)512*KP544*2);
constexpr size_t OFF_WMN   = AL(OFF_WMS   + (size_t)NP_MASK*KP288*2);
constexpr size_t OFF_GRUW  = AL(OFF_WMN   + (size_t)NP_MASK*KP288*2);
constexpr size_t OFF_GI1   = AL(OFF_GRUW  + (size_t)2*3*GRU_NWG*GRU_ROWS*KP288*2);
constexpr size_t OFF_ESIN  = AL(OFF_GI1   + (size_t)2*4*NGATE*4);
constexpr size_t OFF_H1    = AL(OFF_ESIN  + (size_t)2*4*NH*4);
constexpr size_t OFF_H2    = AL(OFF_H1    + (size_t)2*4*NH*4);
constexpr size_t OFF_BAR   = AL(OFF_H2    + (size_t)2*4*NH*4);
constexpr size_t OFF_FR    = AL(OFF_BAR   + 256);
constexpr size_t OFF_DOUT  = AL(OFF_FR    + (size_t)MP*512*2);
constexpr size_t OFF_ES    = AL(OFF_DOUT  + (size_t)MP*NP_DFT*4);
constexpr size_t OFF_EN    = AL(OFF_ES    + (size_t)MP*KP288*2);
constexpr size_t OFF_SR    = AL(OFF_EN    + (size_t)MP*KP288*2);
constexpr size_t OFF_NR    = AL(OFF_SR    + (size_t)MP*NP_MASK*4);
constexpr size_t OFF_VS    = AL(OFF_NR    + (size_t)MP*NP_MASK*4);
constexpr size_t OFF_AID   = AL(OFF_VS    + (size_t)MP*4);
constexpr size_t OFF_FOUT  = AL(OFF_AID   + (size_t)MP*KP544*2);
constexpr size_t OFF_OLA   = AL(OFF_FOUT  + (size_t)MP*512*4);
constexpr size_t WS_TOTAL  = AL(OFF_OLA   + (size_t)BATCH*TOTLEN*4);

__device__ __forceinline__ float sigm(float x) { return 1.f / (1.f + expf(-x)); }

// ---------------------------------------------------------------- zero ws
__global__ void k_zero(unsigned long long* p, size_t n8) {
  size_t i = (size_t)blockIdx.x * blockDim.x + threadIdx.x;
  if (i < n8) p[i] = 0ull;
}

// ---------------------------------------------------------------- tables
// fwd DFT (windowed) rows = [re(257) | im(257)] x 512 time; inv DFT rows = 512 time x 514
__global__ void k_tables(float* win, _Float16* dftT, _Float16* idftT) {
  int gid = blockIdx.x * blockDim.x + threadIdx.x;
  const float w0 = 2.f * PI_F / 512.f;
  if (gid < 512) win[gid] = 0.5f * (1.f - cosf(w0 * gid));
  if (gid < 514 * 512) {
    int j = gid / 512, n = gid % 512;
    float wn = 0.5f * (1.f - cosf(w0 * n));
    float v;
    if (j < NH) { int p = (j * n) & 511;        v =  cosf(w0 * p) * wn; }
    else        { int p = ((j - NH) * n) & 511; v = -sinf(w0 * p) * wn; }
    dftT[j * 512 + n] = (_Float16)v;
  } else if (gid < 514 * 512 + 512 * 514) {
    int r = gid - 514 * 512;
    int n = r / 514, f = r % 514;
    float wn = 0.5f * (1.f - cosf(w0 * n));
    float v;
    if (f < NH) {
      float sc = (f == 0 || f == 256) ? 1.f : 2.f;
      int p = (f * n) & 511;
      v = sc * (1.f / 512.f) * cosf(w0 * p) * wn;
    } else {
      int p = ((f - NH) * n) & 511;
      v = -2.f * (1.f / 512.f) * sinf(w0 * p) * wn;
    }
    idftT[n * KP544 + f] = (_Float16)v;
  }
}

// ---------------------------------------------------------------- mask weight gather (channel-0 rows only: 4f)
__global__ void k_wmask(const float* smW, const float* nmW, _Float16* WS_, _Float16* WN_) {
  int gid = blockIdx.x * blockDim.x + threadIdx.x;
  if (gid >= 2 * NH * NH) return;
  int sel = gid / (NH * NH), rem = gid % (NH * NH);
  int f = rem / NH, k = rem % NH;
  const float* src = sel ? nmW : smW;
  _Float16*    dst = sel ? WN_ : WS_;
  dst[f * KP288 + k] = (_Float16)src[(4 * f) * NH + k];
}

// ---------------------------------------------------------------- GRU weight reorder -> per-WG LDS slabs (f16)
// slab layout: (((ch*3+mat)*16+g)*64 + p)*288 + k ; row p <-> (j = g*17 + p/3, gate = p%3)
__global__ void k_gruw(const float* Whh0_s, const float* Wih1_s, const float* Whh1_s,
                       const float* Whh0_n, const float* Wih1_n, const float* Whh1_n,
                       _Float16* gruW) {
  size_t gid = (size_t)blockIdx.x * blockDim.x + threadIdx.x;
  size_t tot = (size_t)2 * 3 * GRU_NWG * GRU_ROWS * KP288;
  if (gid >= tot) return;
  int k = gid % KP288; size_t rem = gid / KP288;
  int p = rem % GRU_ROWS; rem /= GRU_ROWS;
  int g = rem % GRU_NWG;  rem /= GRU_NWG;
  int mat = rem % 3;      int ch = rem / 3;
  int jl = p / 3, gate = p % 3, j = g * GRU_JSL + jl;
  float v = 0.f;
  if (p < 3 * GRU_JSL && j < NH && k < NH) {
    const float* W = ch ? (mat == 0 ? Whh0_n : mat == 1 ? Wih1_n : Whh1_n)
                        : (mat == 0 ? Whh0_s : mat == 1 ? Wih1_s : Whh1_s);
    v = W[(size_t)(gate * NH + j) * NH + k];
  }
  gruW[gid] = (_Float16)v;
}

// ---------------------------------------------------------------- steering vec + encoder (single row!) + attractors
__global__ __launch_bounds__(256) void k_prep(
    const float* angle, const float* mic,
    const float* W0, const float* b0, const float* Wh, const float* bh,
    const float* Wout, const float* bout,
    const float* ZsW, const float* Zsb, const float* ZnW, const float* Znb,
    float* esin /* [2][4][257] */) {
  __shared__ float sIn[2056];
  __shared__ float sH0[1024];
  __shared__ float sH1[1024];
  __shared__ float sE[NH];
  __shared__ float sTd[4];
  int tid = threadIdx.x;
  if (tid == 0) {
    float td[4];
    float lz = cosf(PI_F * 0.5f);          // ~0 (DIST=1)
    for (int c = 0; c < 4; ++c) {
      float s = 0.f;
      for (int b = 0; b < 4; ++b) {
        float a = -angle[b] * (PI_F / 180.f);
        float lx = cosf(a), ly = sinf(a);  // sin(90deg)=1
        float dx = mic[(b * 4 + c) * 3 + 0] - lx;
        float dy = mic[(b * 4 + c) * 3 + 1] - ly;
        float dz = mic[(b * 4 + c) * 3 + 2] - lz;
        s += dx * dx + dy * dy + dz * dz;
      }
      td[c] = sqrtf(s);
    }
    for (int c = 3; c >= 0; --c) sTd[c] = td[c] - td[0];
  }
  __syncthreads();
  const float KC = 2.f * PI_F * 16000.f / (512.f * 340.4f);
  for (int f = tid; f < NH; f += 256) {
    float re[4], im[4], nn = 0.f;
    for (int c = 0; c < 4; ++c) {
      float ph = -KC * sTd[c] * (float)f;
      re[c] = cosf(ph); im[c] = sinf(ph);
      nn += re[c] * re[c] + im[c] * im[c];
    }
    float inv = rsqrtf(nn);
    for (int c = 0; c < 4; ++c) {
      sIn[c * NH + f]       = re[c] * inv;
      sIn[(4 + c) * NH + f] = im[c] * inv;
    }
  }
  __syncthreads();
  for (int j = tid; j < 1024; j += 256) {
    const float* w = W0 + (size_t)j * 2056;
    float s = b0[j];
    for (int k = 0; k < 2056; ++k) s += w[k] * sIn[k];
    sH0[j] = fmaxf(s, 0.f);
  }
  __syncthreads();
  float* cur = sH0; float* nxt = sH1;
  for (int l = 0; l < 4; ++l) {
    const float* W = Wh + (size_t)l * 1024 * 1024;
    const float* bb = bh + l * 1024;
    for (int j = tid; j < 1024; j += 256) {
      const float* w = W + (size_t)j * 1024;
      float s = bb[j];
      for (int k = 0; k < 1024; ++k) s += w[k] * cur[k];
      nxt[j] = fmaxf(s, 0.f);
    }
    __syncthreads();
    float* t2 = cur; cur = nxt; nxt = t2;
  }
  for (int j = tid; j < NH; j += 256) {
    const float* w = Wout + (size_t)j * 1024;
    float s = bout[j];
    for (int k = 0; k < 1024; ++k) s += w[k] * cur[k];
    sE[j] = s;
  }
  __syncthreads();
  for (int q = tid; q < 2 * 4 * NH; q += 256) {
    int ch = q / (4 * NH), rem = q % (4 * NH), m = rem / NH, j = rem % NH;
    float a = -angle[m] * (PI_F / 180.f);
    float af0 = sinf(a), af1 = cosf(a);
    const float* ZW = ch ? ZnW : ZsW;
    const float* Zb = ch ? Znb : Zsb;
    float g = sigm(af0 * ZW[j * 2 + 0] + af1 * ZW[j * 2 + 1] + Zb[j]);
    esin[q] = sE[j] * g;
  }
}

// ---------------------------------------------------------------- constant layer-1 input gates
__global__ void k_gi1(const float* esin, const float* Wih0_s, const float* bih0_s,
                      const float* Wih0_n, const float* bih0_n, float* gi1) {
  int gid = blockIdx.x * blockDim.x + threadIdx.x;
  if (gid >= 2 * 4 * NGATE) return;
  int ch = gid / (4 * NGATE), rem = gid % (4 * NGATE), m = rem / NGATE, gr = rem % NGATE;
  const float* W  = ch ? Wih0_n : Wih0_s;
  const float* bb = ch ? bih0_n : bih0_s;
  const float* x  = esin + (ch * 4 + m) * NH;
  const float* w  = W + (size_t)gr * NH;
  float s = bb[gr];
  for (int k = 0; k < NH; ++k) s += w[k] * x[k];
  gi1[(ch * 4 + m) * NGATE + gr] = s;
}

// ---------------------------------------------------------------- framing (channel 0, reflect pad, windowed, f16)
__global__ void k_frames(const float* x, const float* win, _Float16* frames) {
  int gid = blockIdx.x * blockDim.x + threadIdx.x;
  if (gid >= BATCH * NT_ * 512) return;
  int n = gid & 511, rem = gid >> 9;
  int t = rem % NT_, b = rem / NT_;
  int idx = t * HOP + n - 256;
  if (idx < 0) idx = -idx;
  if (idx >= LS) idx = 2 * LS - 2 - idx;
  float v = x[(size_t)b * 4 * LS + idx] * win[n];
  frames[(size_t)(b * NT_ + t) * 512 + n] = (_Float16)v;
}

// ---------------------------------------------------------------- generic WMMA GEMM: C[Mp,Np] = A[Mp,Kp] * B[Np,Kp]^T
__global__ __launch_bounds__(256) void k_gemm(const _Float16* __restrict__ A,
                                              const _Float16* __restrict__ Bm,
                                              float* __restrict__ Cm, int Np, int Kp) {
  const int lane = threadIdx.x & 31, wv = threadIdx.x >> 5;
  const int m0 = (blockIdx.y * 2 + (wv >> 2)) * 16;
  const int n0 = (blockIdx.x * 4 + (wv & 3)) * 16;
  const _Float16* ar = A  + (size_t)(m0 + (lane & 15)) * Kp;
  const _Float16* br = Bm + (size_t)(n0 + (lane & 15)) * Kp;
  const int base = (lane < 16) ? 0 : 8;
  v8f acc = {};
  for (int kb = 0; kb < Kp; kb += 32) {
    __builtin_prefetch(br + kb + 256, 0, 1);
    v8h alo = *(const v8h*)(ar + kb + base);
    v8h ahi = *(const v8h*)(ar + kb + 16 + base);
    v8h blo = *(const v8h*)(br + kb + base);
    v8h bhi = *(const v8h*)(br + kb + 16 + base);
    v16h av = __builtin_shufflevector(alo, ahi, 0,1,2,3,4,5,6,7,8,9,10,11,12,13,14,15);
    v16h bv = __builtin_shufflevector(blo, bhi, 0,1,2,3,4,5,6,7,8,9,10,11,12,13,14,15);
    acc = __builtin_amdgcn_wmma_f32_16x16x32_f16(false, av, false, bv, (short)0, acc, false, false);
  }
  const int rowb = m0 + ((lane < 16) ? 0 : 8);
  const int col  = n0 + (lane & 15);
#pragma unroll
  for (int r = 0; r < 8; ++r) Cm[(size_t)(rowb + r) * Np + col] = acc[r];
}

// ---------------------------------------------------------------- GRU persistent kernel
__device__ __forceinline__ void wave_mm(const _Float16* sA, const _Float16* sW,
                                        int nt, int lane, float* outRow) {
  const _Float16* ar = sA + (lane & 15) * KP288;
  const _Float16* br = sW + (nt * 16 + (lane & 15)) * KP288;
  const int base = (lane < 16) ? 0 : 8;
  v8f acc = {};
#pragma unroll
  for (int kt = 0; kt < GRU_KT; ++kt) {
    const int kb = kt * 32;
    v8h alo = *(const v8h*)(ar + kb + base);
    v8h ahi = *(const v8h*)(ar + kb + 16 + base);
    v8h blo = *(const v8h*)(br + kb + base);
    v8h bhi = *(const v8h*)(br + kb + 16 + base);
    v16h av = __builtin_shufflevector(alo, ahi, 0,1,2,3,4,5,6,7,8,9,10,11,12,13,14,15);
    v16h bv = __builtin_shufflevector(blo, bhi, 0,1,2,3,4,5,6,7,8,9,10,11,12,13,14,15);
    acc = __builtin_amdgcn_wmma_f32_16x16x32_f16(false, av, false, bv, (short)0, acc, false, false);
  }
  if (lane < 16) {
#pragma unroll
    for (int r = 0; r < 4; ++r) outRow[r * GRU_ROWS + nt * 16 + lane] = acc[r];
  }
}

__device__ __forceinline__ void chain_barrier(unsigned* c, unsigned target) {
  __threadfence();
  __syncthreads();
  if (threadIdx.x == 0) {
    atomicAdd(c, 1u);
    while (atomicAdd(c, 0u) < target) __builtin_amdgcn_s_sleep(2);
  }
  __syncthreads();
  __threadfence();
}

__global__ __launch_bounds__(256) void k_gru(
    const _Float16* __restrict__ gruW, const float* __restrict__ gi1,
    const float* bhh0_s, const float* bhh0_n,
    const float* bih1_s, const float* bhh1_s,
    const float* bih1_n, const float* bhh1_n,
    float* h1, float* h2, _Float16* es, _Float16* en, unsigned* bar) {
  __shared__ __align__(16) _Float16 sW1 [GRU_ROWS * KP288];
  __shared__ __align__(16) _Float16 sW2i[GRU_ROWS * KP288];
  __shared__ __align__(16) _Float16 sW2h[GRU_ROWS * KP288];
  __shared__ __align__(16) _Float16 sA0 [16 * KP288];
  __shared__ __align__(16) _Float16 sA1 [16 * KP288];
  __shared__ float sGH[4 * GRU_ROWS];
  __shared__ float sGI[4 * GRU_ROWS];

  const int tid = threadIdx.x;
  const int lane = tid & 31, wv = tid >> 5;
  const int ch = blockIdx.x >> 4;
  const int g  = blockIdx.x & 15;
  const int jg0 = g * GRU_JSL;
  const int jcnt = (NH - jg0 < GRU_JSL) ? (NH - jg0) : GRU_JSL;

  // stage weight slabs in LDS via CDNA5 async global->LDS DMA (ASYNCcnt-tracked)
  for (int mat = 0; mat < 3; ++mat) {
    const uint4* s4 = (const uint4*)(gruW + ((size_t)((ch * 3 + mat) * GRU_NWG + g)) * (GRU_ROWS * KP288));
    uint4* d4 = (uint4*)(mat == 0 ? sW1 : mat == 1 ? sW2i : sW2h);
    for (int i = tid; i < GRU_ROWS * KP288 / 8; i += 256) {
      unsigned      ldsa = (unsigned)(size_t)(void*)(d4 + i);  // low 32 bits = LDS offset
      const uint4*  ga   = s4 + i;
      asm volatile("global_load_async_to_lds_b128 %0, %1, off"
                   :: "v"(ldsa), "v"(ga) : "memory");
    }
  }
  {
    uint4 z; z.x = z.y = z.z = z.w = 0u;
    for (int i = tid; i < 16 * KP288 / 8; i += 256) {
      ((uint4*)sA0)[i] = z; ((uint4*)sA1)[i] = z;
    }
  }
  asm volatile("s_wait_asynccnt 0x0" ::: "memory");
  const float* gi1c = gi1 + ch * 4 * NGATE;
  const float* bh0 = ch ? bhh0_n : bhh0_s;
  const float* bi1 = ch ? bih1_n : bih1_s;
  const float* bh1 = ch ? bhh1_n : bhh1_s;
  float* h1c = h1 + ch * 4 * NH;
  float* h2c = h2 + ch * 4 * NH;
  _Float16* eo = ch ? en : es;
  unsigned* cnt = bar + ch;
  unsigned bt = 0;
  __syncthreads();

  for (int t = 0; t < NT_; ++t) {
    // ---- layer 1: gh = h1 @ Whh0^T
    if (wv < 4) wave_mm(sA0, sW1, wv, lane, sGH);
    __syncthreads();
    if (tid < 4 * GRU_JSL) {
      int m = tid / GRU_JSL, jl = tid % GRU_JSL;
      if (jl < jcnt) {
        int j = jg0 + jl, c0 = 3 * jl;
        float ghr = sGH[m * GRU_ROWS + c0]     + bh0[j];
        float ghz = sGH[m * GRU_ROWS + c0 + 1] + bh0[NH + j];
        float ghn = sGH[m * GRU_ROWS + c0 + 2] + bh0[2 * NH + j];
        const float* gim = gi1c + m * NGATE;
        float r = sigm(gim[j] + ghr);
        float z = sigm(gim[NH + j] + ghz);
        float n = tanhf(gim[2 * NH + j] + r * ghn);
        float hp = h1c[m * NH + j];
        h1c[m * NH + j] = (1.f - z) * n + z * hp;
      }
    }
    bt += GRU_NWG; chain_barrier(cnt, bt);
    for (int q = tid; q < 4 * NH; q += 256) {
      int m = q / NH, j = q % NH;
      sA0[m * KP288 + j] = (_Float16)h1c[m * NH + j];
    }
    __syncthreads();
    // ---- layer 2: gi = y1 @ Wih1^T (waves 0-3), gh = h2 @ Whh1^T (waves 4-7)
    if (wv < 4) wave_mm(sA0, sW2i, wv, lane, sGI);
    else        wave_mm(sA1, sW2h, wv - 4, lane, sGH);
    __syncthreads();
    if (tid < 4 * GRU_JSL) {
      int m = tid / GRU_JSL, jl = tid % GRU_JSL;
      if (jl < jcnt) {
        int j = jg0 + jl, c0 = 3 * jl;
        float gir = sGI[m * GRU_ROWS + c0]     + bi1[j];
        float giz = sGI[m * GRU_ROWS + c0 + 1] + bi1[NH + j];
        float gin = sGI[m * GRU_ROWS + c0 + 2] + bi1[2 * NH + j];
        float ghr = sGH[m * GRU_ROWS + c0]     + bh1[j];
        float ghz = sGH[m * GRU_ROWS + c0 + 1] + bh1[NH + j];
        float ghn = sGH[m * GRU_ROWS + c0 + 2] + bh1[2 * NH + j];
        float r = sigm(gir + ghr);
        float z = sigm(giz + ghz);
        float n = tanhf(gin + r * ghn);
        float hp = h2c[m * NH + j];
        float hn_ = (1.f - z) * n + z * hp;
        h2c[m * NH + j] = hn_;
        eo[(size_t)(m * NT_ + t) * KP288 + j] = (_Float16)hn_;
      }
    }
    bt += GRU_NWG; chain_barrier(cnt, bt);
    for (int q = tid; q < 4 * NH; q += 256) {
      int m = q / NH, j = q % NH;
      sA1[m * KP288 + j] = (_Float16)h2c[m * NH + j];
    }
    __syncthreads();
  }
}

// ---------------------------------------------------------------- v_s head
__global__ void k_vs(const _Float16* es, const float* saW, const float* sab, float* vs) {
  int row = blockIdx.x * blockDim.x + threadIdx.x;
  if (row >= MBT) return;
  const _Float16* e = es + (size_t)row * KP288;
  float s = sab[0];
  for (int k = 0; k < NH; ++k) s += (float)e[k] * saW[k];
  vs[row] = sigm(s);
}

// ---------------------------------------------------------------- masks -> w -> Y = w * X  (build iDFT GEMM input)
__global__ void k_apply(const float* Sraw, const float* Nraw, const float* vs,
                        const float* smb, const float* nmb,
                        const float* Dout, _Float16* Aid) {
  int gid = blockIdx.x * blockDim.x + threadIdx.x;
  if (gid >= MBT * NH) return;
  int row = gid / NH, f = gid % NH;
  float S  = tanhf(Sraw[(size_t)row * NP_MASK + f] + smb[4 * f]);
  float Nv = tanhf(Nraw[(size_t)row * NP_MASK + f] + nmb[4 * f]);
  float w  = (S - Nv) * vs[row];
  Aid[(size_t)row * KP544 + f]      = (_Float16)(w * Dout[(size_t)row * NP_DFT + f]);
  Aid[(size_t)row * KP544 + NH + f] = (_Float16)(w * Dout[(size_t)row * NP_DFT + NH + f]);
}

// ---------------------------------------------------------------- overlap-add + normalize
__global__ void k_ola(const float* Fout, float* ola) {
  int gid = blockIdx.x * blockDim.x + threadIdx.x;
  if (gid >= BATCH * NT_ * 512) return;
  int n = gid & 511, rem = gid >> 9;
  int t = rem % NT_, b = rem / NT_;
  atomicAdd(&ola[(size_t)b * TOTLEN + t * HOP + n],
            Fout[(size_t)(b * NT_ + t) * 512 + n]);
}

__global__ void k_norm(const float* ola, const float* win, float* out) {
  int gid = blockIdx.x * blockDim.x + threadIdx.x;
  if (gid >= BATCH * LS) return;
  int b = gid / LS, i = gid % LS;
  int pos = i + 256;
  float ws_ = 0.f;
  int t0 = pos >> 7;
  for (int dt = 0; dt < 4; ++dt) {
    int t = t0 - dt;
    if (t < 0 || t >= NT_) continue;
    int off = pos - t * HOP;
    if (off < 512) { float w = win[off]; ws_ += w * w; }
  }
  out[gid] = ola[(size_t)b * TOTLEN + pos] / ws_;
}

// ---------------------------------------------------------------- launcher
extern "C" void kernel_launch(void* const* d_in, const int* in_sizes, int n_in,
                              void* d_out, int out_size, void* d_ws, size_t ws_size,
                              hipStream_t stream) {
  (void)in_sizes; (void)n_in; (void)out_size; (void)ws_size;
  const float* x      = (const float*)d_in[0];
  const float* angle  = (const float*)d_in[1];
  const float* micpos = (const float*)d_in[2];
  const float* D_W0   = (const float*)d_in[3];
  const float* D_b0   = (const float*)d_in[4];
  const float* D_Wh   = (const float*)d_in[5];
  const float* D_bh   = (const float*)d_in[6];
  const float* D_Wout = (const float*)d_in[7];
  const float* D_bout = (const float*)d_in[8];
  const float* Zs_W   = (const float*)d_in[9];
  const float* Zs_b   = (const float*)d_in[10];
  const float* Zn_W   = (const float*)d_in[11];
  const float* Zn_b   = (const float*)d_in[12];
  const float* sWih0  = (const float*)d_in[13];
  const float* sbih0  = (const float*)d_in[14];
  const float* sWhh0  = (const float*)d_in[15];
  const float* sbhh0  = (const float*)d_in[16];
  const float* sWih1  = (const float*)d_in[17];
  const float* sbih1  = (const float*)d_in[18];
  const float* sWhh1  = (const float*)d_in[19];
  const float* sbhh1  = (const float*)d_in[20];
  const float* nWih0  = (const float*)d_in[21];
  const float* nbih0  = (const float*)d_in[22];
  const float* nWhh0  = (const float*)d_in[23];
  const float* nbhh0  = (const float*)d_in[24];
  const float* nWih1  = (const float*)d_in[25];
  const float* nbih1  = (const float*)d_in[26];
  const float* nWhh1  = (const float*)d_in[27];
  const float* nbhh1  = (const float*)d_in[28];
  const float* smW    = (const float*)d_in[29];
  const float* smb    = (const float*)d_in[30];
  const float* saW    = (const float*)d_in[31];
  const float* sab    = (const float*)d_in[32];
  const float* nmW    = (const float*)d_in[33];
  const float* nmb    = (const float*)d_in[34];

  char* ws = (char*)d_ws;
  float*    win   = (float*)   (ws + OFF_WIN);
  _Float16* dftT  = (_Float16*)(ws + OFF_DFTT);
  _Float16* idftT = (_Float16*)(ws + OFF_IDFTT);
  _Float16* WmS   = (_Float16*)(ws + OFF_WMS);
  _Float16* WmN   = (_Float16*)(ws + OFF_WMN);
  _Float16* gruW  = (_Float16*)(ws + OFF_GRUW);
  float*    gi1   = (float*)   (ws + OFF_GI1);
  float*    esin  = (float*)   (ws + OFF_ESIN);
  float*    h1    = (float*)   (ws + OFF_H1);
  float*    h2    = (float*)   (ws + OFF_H2);
  unsigned* bar   = (unsigned*)(ws + OFF_BAR);
  _Float16* frames= (_Float16*)(ws + OFF_FR);
  float*    Dout  = (float*)   (ws + OFF_DOUT);
  _Float16* eS    = (_Float16*)(ws + OFF_ES);
  _Float16* eN    = (_Float16*)(ws + OFF_EN);
  float*    Sraw  = (float*)   (ws + OFF_SR);
  float*    Nraw  = (float*)   (ws + OFF_NR);
  float*    vs    = (float*)   (ws + OFF_VS);
  _Float16* Aid   = (_Float16*)(ws + OFF_AID);
  float*    Fout  = (float*)   (ws + OFF_FOUT);
  float*    ola   = (float*)   (ws + OFF_OLA);
  float*    out   = (float*)   d_out;

  auto blocks = [](size_t n, int b) { return (unsigned)((n + b - 1) / b); };

  // 0) zero whole workspace (h state, barrier counter, pads) each launch
  size_t n8 = WS_TOTAL / 8;
  k_zero<<<blocks(n8, 256), 256, 0, stream>>>((unsigned long long*)ws, n8);

  // 1) tables, weight staging
  k_tables<<<blocks(514 * 512 + 512 * 514, 256), 256, 0, stream>>>(win, dftT, idftT);
  k_wmask <<<blocks(2 * NH * NH, 256), 256, 0, stream>>>(smW, nmW, WmS, WmN);
  k_gruw  <<<blocks((size_t)2 * 3 * GRU_NWG * GRU_ROWS * KP288, 256), 256, 0, stream>>>(
      sWhh0, sWih1, sWhh1, nWhh0, nWih1, nWhh1, gruW);

  // 2) steering vector + encoder (single row) + attractor gates
  k_prep<<<1, 256, 0, stream>>>(angle, micpos, D_W0, D_b0, D_Wh, D_bh, D_Wout, D_bout,
                                Zs_W, Zs_b, Zn_W, Zn_b, esin);
  k_gi1<<<blocks(2 * 4 * NGATE, 256), 256, 0, stream>>>(esin, sWih0, sbih0, nWih0, nbih0, gi1);

  // 3) STFT (channel 0) as WMMA GEMM against windowed DFT basis
  k_frames<<<blocks((size_t)BATCH * NT_ * 512, 256), 256, 0, stream>>>(x, win, frames);
  k_gemm<<<dim3(NP_DFT / 64, MP / 32), 256, 0, stream>>>(frames, dftT, Dout, NP_DFT, KP512);

  // 4) persistent GRU: 2 chains x 16 WGs, LDS-resident weights, WMMA per step
  k_gru<<<32, 256, 0, stream>>>(gruW, gi1, sbhh0, nbhh0, sbih1, sbhh1, nbih1, nbhh1,
                                h1, h2, eS, eN, bar);

  // 5) heads: v_s + channel-0 mask GEMMs
  k_vs<<<blocks(MBT, 256), 256, 0, stream>>>(eS, saW, sab, vs);
  k_gemm<<<dim3(NP_MASK / 64, MP / 32), 256, 0, stream>>>(eS, WmS, Sraw, NP_MASK, KP288);
  k_gemm<<<dim3(NP_MASK / 64, MP / 32), 256, 0, stream>>>(eN, WmN, Nraw, NP_MASK, KP288);

  // 6) masking + iSTFT (iDFT GEMM, OLA, normalize)
  k_apply<<<blocks((size_t)MBT * NH, 256), 256, 0, stream>>>(Sraw, Nraw, vs, smb, nmb, Dout, Aid);
  k_gemm<<<dim3(NP_TIME / 64, MP / 32), 256, 0, stream>>>(Aid, idftT, Fout, NP_TIME, KP544);
  k_ola<<<blocks((size_t)BATCH * NT_ * 512, 256), 256, 0, stream>>>(Fout, ola);
  k_norm<<<blocks((size_t)BATCH * LS, 256), 256, 0, stream>>>(ola, win, out);
}
